// ContextAttention_57870389346886
// MI455X (gfx1250) — matrix operand order repaired and checked by reading, not compiled, not run on hardware
//
#include <hip/hip_runtime.h>

typedef __bf16 bf16_t;
typedef __bf16 v16bf __attribute__((ext_vector_type(16)));
typedef float  v8f   __attribute__((ext_vector_type(8)));

#define HH 128      // full-res H=W
#define CC 128      // channels
#define LL 4096     // 64*64 downsampled positions
#define K1 1152     // C*3*3
#define N2 2048     // C*4*4

// ---------------- im2col of downsampled feature (3x3, stride1, pad1) ----------------
// src: one image [C,128,128] full-res (we sample ::2,::2); dst: [L][K1] bf16, k = c*9+ky*3+kx
__global__ void k_im2col_d(const float* __restrict__ src, bf16_t* __restrict__ dst) {
  long t = (long)blockIdx.x * 256 + threadIdx.x;
  if (t >= (long)LL * K1) return;
  int p = (int)(t / K1), k = (int)(t % K1);
  int c = k / 9, r = k % 9, ky = r / 3, kx = r % 3;
  int i = p >> 6, j = p & 63;
  int yy = i + ky - 1, xx = j + kx - 1;
  float v = 0.f;
  if ((unsigned)yy < 64u && (unsigned)xx < 64u)
    v = src[((long)c * HH + 2 * yy) * HH + 2 * xx];
  dst[t] = (bf16_t)v;
}

// ---------------- per-patch L2 norm of bg_d 3x3 patches (f32 precision) ----------------
__global__ void k_norm(const float* __restrict__ bg, float* __restrict__ norm) {
  __shared__ float red[128];
  int l = blockIdx.x;
  int li = l >> 6, lj = l & 63;
  float s = 0.f;
  for (int k = threadIdx.x; k < K1; k += 128) {
    int c = k / 9, r = k % 9, ky = r / 3, kx = r % 3;
    int yy = li + ky - 1, xx = lj + kx - 1;
    if ((unsigned)yy < 64u && (unsigned)xx < 64u) {
      float v = bg[((long)c * HH + 2 * yy) * HH + 2 * xx];
      s += v * v;
    }
  }
  red[threadIdx.x] = s;
  __syncthreads();
  for (int o = 64; o > 0; o >>= 1) {
    if ((int)threadIdx.x < o) red[threadIdx.x] += red[threadIdx.x + o];
    __syncthreads();
  }
  if (threadIdx.x == 0) norm[l] = sqrtf(red[0]);
}

// ---------------- patch validity: 3x3 mask patch sums to zero ----------------
__global__ void k_valid(const float* __restrict__ mask, float* __restrict__ valid) {
  int l = blockIdx.x * 256 + threadIdx.x;
  if (l >= LL) return;
  int li = l >> 6, lj = l & 63;
  float s = 0.f;
  for (int ky = 0; ky < 3; ++ky)
    for (int kx = 0; kx < 3; ++kx) {
      int yy = li + ky - 1, xx = lj + kx - 1;
      if ((unsigned)yy < 64u && (unsigned)xx < 64u) s += mask[(long)(2 * yy) * HH + 2 * xx];
    }
  valid[l] = (s == 0.f) ? 1.f : 0.f;
}

// ---------------- deconv weights transposed: W2t[m=c*16+ky*4+kx][l] (4x4, stride2, pad1) ----------------
__global__ void k_w2t(const float* __restrict__ bg, bf16_t* __restrict__ w2t) {
  long t = (long)blockIdx.x * 256 + threadIdx.x;
  if (t >= (long)N2 * LL) return;
  int m = (int)(t >> 12), l = (int)(t & 4095);
  int c = m >> 4, r = m & 15, ky = r >> 2, kx = r & 3;
  int yy = 2 * (l >> 6) + ky - 1, xx = 2 * (l & 63) + kx - 1;
  float v = 0.f;
  if ((unsigned)yy < 128u && (unsigned)xx < 128u) v = bg[((long)c * HH + yy) * HH + xx];
  w2t[t] = (bf16_t)v;
}

// ---------------- bf16 WMMA GEMM: C[m,n] = sum_k A[m,k]*B[n,k], optional 1/max(norm[n],1e-4) col scale ----
// WG tile 128(M) x 128(N), 8 waves in 2(M) x 4(N), wave tile 64x32, K step 32.
// Staging: gfx1250 GLOBAL_LOAD_ASYNC_TO_LDS_B128 (ASYNCcnt) into a 4-deep LDS pipeline,
// one workgroup barrier per K-step (buffer (kt+2)&3 was last read at kt-2, protected by the kt-1 barrier).
union FragU { v16bf v; unsigned int u[8]; };

#define LDS_HALF (256 * 20)          // one buffer: A(128 rows) + B(128 rows), 20 u32 (80B) row stride
#define LDS_HALF_BYTES 20480u

__global__ __launch_bounds__(256)
void k_gemm(const bf16_t* __restrict__ A, int lda,
            const bf16_t* __restrict__ B, int ldb,
            float* __restrict__ Cm, int ldc, int K,
            const float* __restrict__ colNorm) {
  __shared__ unsigned int lds[4 * LDS_HALF];
  const int tid   = threadIdx.x;
  const int wave  = tid >> 5, lane = tid & 31;
  const int waveM = wave & 1, waveN = wave >> 1;
  const int mlane = lane & 15, half = lane >> 4;
  const long tileM0 = (long)blockIdx.y * 128, tileN0 = (long)blockIdx.x * 128;

  // this thread async-copies four 16B chunks per K-step: A rows {row0,row1}, B rows {row0,row1}
  const int row0 = tid >> 2, q = tid & 3;
  const int row1 = row0 + 64;
  const bf16_t* gA0 = A + (tileM0 + row0) * lda + q * 8;
  const bf16_t* gA1 = A + (tileM0 + row1) * lda + q * 8;
  const bf16_t* gB0 = B + (tileN0 + row0) * ldb + q * 8;
  const bf16_t* gB1 = B + (tileN0 + row1) * ldb + q * 8;
  const unsigned la0 = (unsigned)(unsigned long long)(const void*)&lds[row0 * 20 + q * 4];
  const unsigned la1 = (unsigned)(unsigned long long)(const void*)&lds[row1 * 20 + q * 4];
  const unsigned lb0 = (unsigned)(unsigned long long)(const void*)&lds[(128 + row0) * 20 + q * 4];
  const unsigned lb1 = (unsigned)(unsigned long long)(const void*)&lds[(128 + row1) * 20 + q * 4];

  v8f acc[4][2] = {};
  const int nk = K >> 5;

  auto issue = [&](long ko, int bi) {
    const unsigned bo = (unsigned)bi * LDS_HALF_BYTES;
    asm volatile("global_load_async_to_lds_b128 %0, %1, off" :: "v"(la0 + bo), "v"(gA0 + ko) : "memory");
    asm volatile("global_load_async_to_lds_b128 %0, %1, off" :: "v"(la1 + bo), "v"(gA1 + ko) : "memory");
    asm volatile("global_load_async_to_lds_b128 %0, %1, off" :: "v"(lb0 + bo), "v"(gB0 + ko) : "memory");
    asm volatile("global_load_async_to_lds_b128 %0, %1, off" :: "v"(lb1 + bo), "v"(gB1 + ko) : "memory");
  };

  auto compute = [&](int bi) {
    const unsigned int* ldsA = lds + bi * LDS_HALF;
    const unsigned int* ldsB = ldsA + 128 * 20;
    // gather fragments per CDNA5 16-bit A(16x32)/B(32x16) VGPR layouts
    FragU af[4], bf[2];
    #pragma unroll
    for (int ms = 0; ms < 4; ++ms) {
      int rb = (waveM * 64 + ms * 16 + mlane) * 20;
      #pragma unroll
      for (int j = 0; j < 8; ++j) {
        int pair = (j < 4) ? (j + 4 * half) : (8 + (j - 4) + 4 * half);
        af[ms].u[j] = ldsA[rb + pair];
      }
    }
    #pragma unroll
    for (int ns = 0; ns < 2; ++ns) {
      int rb = (waveN * 32 + ns * 16 + mlane) * 20;
      #pragma unroll
      for (int j = 0; j < 8; ++j) bf[ns].u[j] = ldsB[rb + ((j < 4) ? (j + 4 * half) : (8 + (j - 4) + 4 * half))];
    }
    #pragma unroll
    for (int ms = 0; ms < 4; ++ms)
      #pragma unroll
      for (int ns = 0; ns < 2; ++ns)
        acc[ms][ns] = __builtin_amdgcn_wmma_f32_16x16x32_bf16(
            false, af[ms].v, false, bf[ns].v, (short)0, acc[ms][ns], false, false);
  };

  // prologue: tiles 0 and 1 in flight (nk >= 2 for all our shapes)
  issue(0, 0);
  issue(32, 1);

  // steady state: branch-free; 3 tiles in flight (12 async ops), wait <=8 drains tile kt
  for (int kt = 0; kt < nk - 2; ++kt) {
    issue((long)(kt + 2) * 32, (kt + 2) & 3);
    asm volatile("s_wait_asynccnt 0x8" ::: "memory");
    __builtin_prefetch(gA0 + (long)(kt + 3) * 32, 0, 1);
    __builtin_prefetch(gB0 + (long)(kt + 3) * 32, 0, 1);
    __syncthreads();
    compute(kt & 3);
  }
  // tail: tile nk-2, then tile nk-1
  asm volatile("s_wait_asynccnt 0x4" ::: "memory");
  __syncthreads();
  compute((nk - 2) & 3);
  asm volatile("s_wait_asynccnt 0x0" ::: "memory");
  __syncthreads();
  compute((nk - 1) & 3);

  // epilogue: C/D layout -> m = r + 8*half (+16*ms ...), n = mlane (+16*ns ...)
  #pragma unroll
  for (int ns = 0; ns < 2; ++ns) {
    long n = tileN0 + waveN * 32 + ns * 16 + mlane;
    float scale = 1.f;
    if (colNorm) scale = 1.f / fmaxf(colNorm[n], 1e-4f);
    #pragma unroll
    for (int ms = 0; ms < 4; ++ms) {
      long mbase = tileM0 + waveM * 64 + ms * 16 + half * 8;
      #pragma unroll
      for (int r = 0; r < 8; ++r)
        Cm[(mbase + r) * ldc + n] = acc[ms][ns][r] * scale;
    }
  }
}

// ---------------- diagonal 3-tap blur on flat [L,L] (pass 1) ----------------
__global__ void k_fuse1(const float* __restrict__ in, float* __restrict__ out) {
  long t = (long)blockIdx.x * 256 + threadIdx.x;  // t = p*4096 + l
  int p = (int)(t >> 12), l = (int)(t & 4095);
  float a = 0.f;
  #pragma unroll
  for (int d = -1; d <= 1; ++d) {
    int pp = p + d, ll = l + d;
    if ((unsigned)pp < 4096u && (unsigned)ll < 4096u) a += in[(long)pp * 4096 + ll];
  }
  out[t] = a;
}

__device__ __forceinline__ int swap64(int t) { return ((t & 63) << 6) | (t >> 6); }

// ---------------- diagonal blur in base-64 digit-swapped coordinates (pass 2) ----------------
__global__ void k_fuse2(const float* __restrict__ in, float* __restrict__ out) {
  long t = (long)blockIdx.x * 256 + threadIdx.x;
  int p = (int)(t >> 12), l = (int)(t & 4095);
  int lp = swap64(l), pp = swap64(p);
  float a = 0.f;
  #pragma unroll
  for (int d = -1; d <= 1; ++d) {
    int l2 = lp + d, p2 = pp + d;
    if ((unsigned)l2 < 4096u && (unsigned)p2 < 4096u)
      a += in[(long)swap64(p2) * 4096 + swap64(l2)];
  }
  out[t] = a;
}

// ---------------- masked softmax over l per position p + argmax, emit bf16 attention ----------------
__global__ __launch_bounds__(256)
void k_softmax(const float* __restrict__ corr, const float* __restrict__ valid,
               bf16_t* __restrict__ smOut, float* __restrict__ matched) {
  __shared__ float red[256];
  __shared__ float rv[256];
  __shared__ int   ri[256];
  const int p = blockIdx.x;
  const float* row = corr + (long)p * 4096;
  // max of z = vm * corr / TEMP  (invalid rows contribute z=0, as in reference)
  float lmax = -3.4e38f;
  for (int l = threadIdx.x; l < 4096; l += 256)
    lmax = fmaxf(lmax, valid[l] * row[l] * 10.f);
  red[threadIdx.x] = lmax; __syncthreads();
  for (int o = 128; o > 0; o >>= 1) {
    if ((int)threadIdx.x < o) red[threadIdx.x] = fmaxf(red[threadIdx.x], red[threadIdx.x + o]);
    __syncthreads();
  }
  float mx = red[0]; __syncthreads();
  // sum exp(z - mx) over all l (including invalid zeros)
  float lsum = 0.f;
  for (int l = threadIdx.x; l < 4096; l += 256)
    lsum += __expf(valid[l] * row[l] * 10.f - mx);
  red[threadIdx.x] = lsum; __syncthreads();
  for (int o = 128; o > 0; o >>= 1) {
    if ((int)threadIdx.x < o) red[threadIdx.x] += red[threadIdx.x + o];
    __syncthreads();
  }
  float inv = 1.f / red[0];
  // emit vm*softmax (bf16) and track argmax (first occurrence on ties)
  float bv = -1.f; int bi = 0;
  for (int l = threadIdx.x; l < 4096; l += 256) {
    float vm = valid[l];
    float e = vm * __expf(vm * row[l] * 10.f - mx);
    smOut[(long)p * 4096 + l] = (bf16_t)(e * inv);
    if (e > bv) { bv = e; bi = l; }
  }
  rv[threadIdx.x] = bv; ri[threadIdx.x] = bi; __syncthreads();
  for (int o = 128; o > 0; o >>= 1) {
    if ((int)threadIdx.x < o) {
      float ov = rv[threadIdx.x + o]; int oi = ri[threadIdx.x + o];
      if (ov > rv[threadIdx.x] || (ov == rv[threadIdx.x] && oi < ri[threadIdx.x])) {
        rv[threadIdx.x] = ov; ri[threadIdx.x] = oi;
      }
    }
    __syncthreads();
  }
  if (threadIdx.x == 0) {
    int idx = ri[0];
    matched[p]        = (float)(idx >> 6);
    matched[4096 + p] = (float)(idx & 63);
  }
}

// ---------------- overlap-add of 4x4 patch contributions (transposed conv gather), /4 ----------------
__global__ void k_recon(const float* __restrict__ recp, float* __restrict__ rec) {
  long t = (long)blockIdx.x * 256 + threadIdx.x;  // c*16384 + y*128 + x
  if (t >= (long)CC * HH * HH) return;
  int c = (int)(t >> 14), y = (int)((t >> 7) & 127), x = (int)(t & 127);
  float a = 0.f;
  int i0 = (y + 1) >> 1, j0 = (x + 1) >> 1;
  #pragma unroll
  for (int di = 0; di < 2; ++di) {
    int i = i0 - di, ky = y - 2 * i + 1;
    if ((unsigned)i >= 64u || (unsigned)ky >= 4u) continue;
    #pragma unroll
    for (int dj = 0; dj < 2; ++dj) {
      int j = j0 - dj, kx = x - 2 * j + 1;
      if ((unsigned)j >= 64u || (unsigned)kx >= 4u) continue;
      a += recp[(long)(i * 64 + j) * 2048 + c * 16 + ky * 4 + kx];
    }
  }
  rec[t] = a * 0.25f;
}

extern "C" void kernel_launch(void* const* d_in, const int* in_sizes, int n_in,
                              void* d_out, int out_size, void* d_ws, size_t ws_size,
                              hipStream_t stream) {
  const float* fg = (const float*)d_in[0];
  const float* bg = (const float*)d_in[1];
  const float* mk = (const float*)d_in[2];
  float* out = (float*)d_out;

  char* ws = (char*)d_ws;
  size_t off = 0;
  auto take = [&](size_t bytes) -> void* {
    void* p = ws + off;
    off = (off + bytes + 255) & ~(size_t)255;
    return p;
  };
  bf16_t* A1    = (bf16_t*)take((size_t)LL * K1 * 2);   // fg_d im2col   [L][K1]
  bf16_t* B1    = (bf16_t*)take((size_t)LL * K1 * 2);   // bg_d im2col   [L][K1]
  float*  nrm   = (float*) take((size_t)LL * 4);
  float*  vld   = (float*) take((size_t)LL * 4);
  float*  corrA = (float*) take((size_t)LL * LL * 4);   // [p][l]
  float*  corrB = (float*) take((size_t)LL * LL * 4);
  bf16_t* A2    = (bf16_t*)take((size_t)LL * LL * 2);   // softmaxed attention, bf16
  bf16_t* W2t   = (bf16_t*)take((size_t)N2 * LL * 2);   // deconv weights [m][l]
  float*  recp  = (float*) take((size_t)LL * N2 * 4);   // patch reconstructions [p][m]
  (void)ws_size; (void)in_sizes; (void)n_in; (void)out_size;

  const long REC_TOT = (long)4 * CC * HH * HH;          // 8,388,608 floats

  for (int n = 0; n < 4; ++n) {
    const float* imgFG = fg + (long)n * CC * HH * HH;
    const float* imgBG = bg + (long)n * CC * HH * HH;
    const float* imgMK = mk + (long)n * HH * HH;

    k_im2col_d<<<(LL * K1 + 255) / 256, 256, 0, stream>>>(imgFG, A1);
    k_im2col_d<<<(LL * K1 + 255) / 256, 256, 0, stream>>>(imgBG, B1);
    k_norm<<<LL, 128, 0, stream>>>(imgBG, nrm);
    k_valid<<<LL / 256, 256, 0, stream>>>(imgMK, vld);
    k_w2t<<<((long)N2 * LL + 255) / 256, 256, 0, stream>>>(imgBG, W2t);

    // GEMM1: corr[p][l], M=N=4096, K=1152, col scale 1/max(norm,1e-4)
    k_gemm<<<dim3(LL / 128, LL / 128), 256, 0, stream>>>(A1, K1, B1, K1, corrA, LL, K1, nrm);

    k_fuse1<<<(int)(((long)LL * LL) / 256), 256, 0, stream>>>(corrA, corrB);
    k_fuse2<<<(int)(((long)LL * LL) / 256), 256, 0, stream>>>(corrB, corrA);

    k_softmax<<<LL, 256, 0, stream>>>(corrA, vld, A2, out + REC_TOT + (long)n * 2 * LL);

    // GEMM2: recp[p][m], M=4096, N=2048, K=4096
    k_gemm<<<dim3(N2 / 128, LL / 128), 256, 0, stream>>>(A2, LL, W2t, LL, recp, N2, LL, nullptr);

    k_recon<<<(int)(((long)CC * HH * HH + 255) / 256), 256, 0, stream>>>(recp, out + (long)n * CC * HH * HH);
  }
}
